// Block_30262339567868
// MI455X (gfx1250) — matrix operand order repaired and checked by reading, not compiled, not run on hardware
//
#include <hip/hip_runtime.h>
#include <hip/hip_bf16.h>

// ---------------------------------------------------------------------------
// Types for CDNA5 WMMA (wave32): D(16x16 f32) = A(16x32 bf16) x B(32x16 bf16) + C
// ---------------------------------------------------------------------------
typedef __bf16 bf16;
typedef __attribute__((ext_vector_type(16))) __bf16 v16bf;
typedef __attribute__((ext_vector_type(8)))  float  v8f;
typedef __attribute__((ext_vector_type(4)))  unsigned uint4v;
typedef __attribute__((ext_vector_type(8)))  int    int8v;
typedef __attribute__((ext_vector_type(4)))  int    int4v;

#define KB 256   // K-chunk staged in LDS per workgroup

__device__ __forceinline__ v8f wmma_bf16(v16bf a, v16bf b, v8f c) {
    return __builtin_amdgcn_wmma_f32_16x16x32_bf16(
        false, a, false, b, (short)0, c, false, false);
}

// ---------------------------------------------------------------------------
// Tensor Data Mover: stage a 2-D tile (tile_d1 rows x tile_d0 elems) into LDS.
// Descriptor per CDNA5 ISA ch.8: group0 = {count/lds_addr/global_addr/type},
// group1 = {data_size, tensor dims, tile dims, dim0 stride}. 2-D tensor ->
// groups 2/3 zero. Tracked by TENSORcnt. Issued by one wave (EXEC ignored).
// ---------------------------------------------------------------------------
#if __has_builtin(__builtin_amdgcn_tensor_load_to_lds)
#define HAVE_TDM 1
#endif

__device__ __forceinline__ void tdm_load_tile(void* lds_ptr, const void* gptr,
                                              unsigned elem_log2,      // 1=2B, 2=4B
                                              unsigned tile_d0,        // elems per row
                                              unsigned tile_d1,        // rows
                                              unsigned long long d0_stride) { // elems
#if defined(HAVE_TDM)
    unsigned lds_off = (unsigned)(uintptr_t)lds_ptr;   // LDS aperture: low 32 = offset
    unsigned long long ga = (unsigned long long)(uintptr_t)gptr;
    uint4v g0;
    g0[0] = 1u;                                     // count=1 valid descriptor
    g0[1] = lds_off;                                // lds_addr (bytes)
    g0[2] = (unsigned)ga;                           // global_addr[31:0]
    g0[3] = (unsigned)((ga >> 32) & 0x01ffffffu) | (2u << 30); // addr[56:32]|type=2
    int8v g1;
    g1[0] = (int)(elem_log2 << 16);                 // wg_mask=0, data_size
    g1[1] = (int)((tile_d0 & 0xffffu) << 16);       // tensor_dim0[15:0]
    g1[2] = (int)((tile_d0 >> 16) | ((tile_d1 & 0xffffu) << 16)); // td0 hi | td1 lo
    g1[3] = (int)((tile_d1 >> 16) | (tile_d0 << 16));             // td1 hi | tile_dim0
    g1[4] = (int)(tile_d1 & 0xffffu);               // tile_dim1 (tile_dim2=0)
    g1[5] = (int)(d0_stride & 0xffffffffu);         // tensor_dim0_stride[31:0]
    g1[6] = (int)((d0_stride >> 32) & 0xffffu);     // stride[47:32]
    g1[7] = 0;
    int4v gz = {0, 0, 0, 0};
#if __clang_major__ >= 23
    int8v gz8 = {0, 0, 0, 0, 0, 0, 0, 0};
    __builtin_amdgcn_tensor_load_to_lds(g0, g1, gz, gz, gz8, 0);
#else
    __builtin_amdgcn_tensor_load_to_lds(g0, g1, gz, gz, 0);
#endif
    __builtin_amdgcn_s_wait_tensorcnt(0);
#else
    // Cooperative fallback: the issuing wave copies the tile itself.
    int lane = threadIdx.x & 31;
    if (elem_log2 == 2) {
        float* dst = (float*)lds_ptr;
        const float* src = (const float*)gptr;
        for (unsigned r = 0; r < tile_d1; r++)
            for (unsigned i = lane; i < tile_d0; i += 32)
                dst[r * tile_d0 + i] = src[(size_t)r * d0_stride + i];
    } else {
        bf16* dst = (bf16*)lds_ptr;
        const bf16* src = (const bf16*)gptr;
        for (unsigned r = 0; r < tile_d1; r++)
            for (unsigned i = lane; i < tile_d0; i += 32)
                dst[r * tile_d0 + i] = src[(size_t)r * d0_stride + i];
    }
#endif
}

// ---------------------------------------------------------------------------
// Fragment loaders per ISA 7.12.2 layouts (wave32)
// A (16x32, 16-bit): lane row = lane&15; halves[0..7]=K(kb..kb+7),
//                    halves[8..15]=K(16+kb..), kb = (lane>=16)?8:0
// B (32x16, 16-bit): lane col = lane&15; same K order as A, kb=(lane>=16)?8:0
// C (16x16, f32):    lane col = lane&15; vgpr r -> row r + 8*(lane>=16)
// ---------------------------------------------------------------------------
__device__ __forceinline__ v16bf load_A_lds_f32(const float* sh, int lda, int k0) {
    int lane = threadIdx.x & 31;
    const float* p = sh + (lane & 15) * lda + k0 + ((lane >> 4) << 3);
    v16bf a;
#pragma unroll
    for (int i = 0; i < 8; i++) { a[i] = (bf16)p[i]; a[8 + i] = (bf16)p[16 + i]; }
    return a;
}

__device__ __forceinline__ v16bf load_A_lds_bf16(const bf16* sh, int lda, int k0) {
    int lane = threadIdx.x & 31;
    const bf16* p = sh + (lane & 15) * lda + k0 + ((lane >> 4) << 3);
    v16bf a;
#pragma unroll
    for (int i = 0; i < 8; i++) { a[i] = p[i]; a[8 + i] = p[16 + i]; }
    return a;
}

__device__ __forceinline__ v16bf load_A_bf16(const bf16* A, int lda, int m0, int k0) {
    int lane = threadIdx.x & 31;
    const bf16* p = A + (size_t)(m0 + (lane & 15)) * lda + k0 + ((lane >> 4) << 3);
    v16bf a;
#pragma unroll
    for (int i = 0; i < 8; i++) { a[i] = p[i]; a[8 + i] = p[16 + i]; }
    return a;
}

// B[k][n] = W[n][k] for row-major W[N][K] (computes X @ W^T)
__device__ __forceinline__ v16bf load_BT_f32(const float* W, int ldw, int n0, int k0) {
    int lane = threadIdx.x & 31;
    const float* p = W + (size_t)(n0 + (lane & 15)) * ldw + k0 + ((lane >> 4) << 3);
    v16bf b;
#pragma unroll
    for (int i = 0; i < 8; i++) { b[i] = (bf16)p[i]; b[8 + i] = (bf16)p[16 + i]; }
    return b;
}

__device__ __forceinline__ v16bf load_BT_bf16(const bf16* W, int ldw, int n0, int k0) {
    int lane = threadIdx.x & 31;
    const bf16* p = W + (size_t)(n0 + (lane & 15)) * ldw + k0 + ((lane >> 4) << 3);
    v16bf b;
#pragma unroll
    for (int i = 0; i < 8; i++) { b[i] = p[i]; b[8 + i] = p[16 + i]; }
    return b;
}

__device__ __forceinline__ float wave_red_sum(float v) {
#pragma unroll
    for (int o = 16; o > 0; o >>= 1) v += __shfl_xor(v, o, 32);
    return v;
}

// ---------------------------------------------------------------------------
// RMSNorm: one block per token
// ---------------------------------------------------------------------------
__global__ void k_rmsnorm(const float* __restrict__ x, const float* __restrict__ w,
                          float* __restrict__ out, int C) {
    int t = blockIdx.x;
    const float* xr = x + (size_t)t * C;
    float ss = 0.f;
    for (int i = threadIdx.x; i < C; i += blockDim.x) { float v = xr[i]; ss += v * v; }
    __shared__ float sbuf[32];
    ss = wave_red_sum(ss);
    int wid = threadIdx.x >> 5, lid = threadIdx.x & 31;
    if (lid == 0) sbuf[wid] = ss;
    __syncthreads();
    int nw = blockDim.x >> 5;
    float tot = (threadIdx.x < (unsigned)nw) ? sbuf[threadIdx.x] : 0.f;
    tot = wave_red_sum(tot);
    if (threadIdx.x == 0) sbuf[0] = tot;
    __syncthreads();
    float rms = rsqrtf(sbuf[0] / (float)C + 1e-6f);
    for (int i = threadIdx.x; i < C; i += blockDim.x)
        out[(size_t)t * C + i] = xr[i] * rms * w[i];
}

// ---------------------------------------------------------------------------
// GEMM: C[M,N] = A[M,K](f32) @ W[N,K]^T (+ res). TDM stages a 32xKB A-chunk
// into LDS once per block; each of 4 waves computes a 32x64 strip: every
// B-fragment load feeds 2 wmmas. grid = (M/32, N/256), block = (32,4).
// ---------------------------------------------------------------------------
__global__ void k_gemm_xwT(const float* __restrict__ A, const float* __restrict__ W,
                           const float* __restrict__ res, float* __restrict__ C,
                           int M, int N, int K) {
    __shared__ float Ash[32 * KB];
    int m0 = blockIdx.x * 32;
    int n0 = (blockIdx.y * blockDim.y + threadIdx.y) * 64;
    v8f acc0[4] = {{}, {}, {}, {}};
    v8f acc1[4] = {{}, {}, {}, {}};
    for (int kc = 0; kc < K; kc += KB) {
        if (threadIdx.y == 0)
            tdm_load_tile(Ash, A + (size_t)m0 * K + kc, 2u, KB, 32u,
                          (unsigned long long)K);
        __syncthreads();
        for (int k0 = 0; k0 < KB; k0 += 32) {
            __builtin_prefetch(W + (size_t)n0 * K + kc + k0 + 64, 0, 1);
            v16bf a0 = load_A_lds_f32(Ash, KB, k0);
            v16bf a1 = load_A_lds_f32(Ash + 16 * KB, KB, k0);
#pragma unroll
            for (int j = 0; j < 4; j++) {
                v16bf b = load_BT_f32(W, K, n0 + j * 16, kc + k0);
                acc0[j] = wmma_bf16(a0, b, acc0[j]);
                acc1[j] = wmma_bf16(a1, b, acc1[j]);
            }
        }
        __syncthreads();
    }
    int lane = threadIdx.x & 31;
    int cn = lane & 15, mh = (lane >> 4) << 3;
    if (res != nullptr) {
#pragma unroll
        for (int j = 0; j < 4; j++) {
            int n = n0 + j * 16 + cn;
            float old[8];
#pragma unroll
            for (int r = 0; r < 8; r++) old[r] = res[(size_t)(m0 + mh + r) * N + n];
#pragma unroll
            for (int r = 0; r < 8; r++)
                C[(size_t)(m0 + mh + r) * N + n] = acc0[j][r] + old[r];
#pragma unroll
            for (int r = 0; r < 8; r++) old[r] = res[(size_t)(m0 + 16 + mh + r) * N + n];
#pragma unroll
            for (int r = 0; r < 8; r++)
                C[(size_t)(m0 + 16 + mh + r) * N + n] = acc1[j][r] + old[r];
        }
    } else {
#pragma unroll
        for (int j = 0; j < 4; j++) {
            int n = n0 + j * 16 + cn;
#pragma unroll
            for (int r = 0; r < 8; r++)
                C[(size_t)(m0 + mh + r) * N + n] = acc0[j][r];
#pragma unroll
            for (int r = 0; r < 8; r++)
                C[(size_t)(m0 + 16 + mh + r) * N + n] = acc1[j][r];
        }
    }
}

// ---------------------------------------------------------------------------
// RoPE + head split + bf16 pack. qkv[T,3072] -> qb/kb [H][T][64], vt [H][64][T]
// ---------------------------------------------------------------------------
__global__ void k_rope_pack(const float* __restrict__ qkv, bf16* __restrict__ qb,
                            bf16* __restrict__ kb, bf16* __restrict__ vt, int T) {
    int t = blockIdx.x;
    const float* row = qkv + (size_t)t * 3072;
    for (int idx = threadIdx.x; idx < 512; idx += blockDim.x) {
        int h = idx >> 5, j = idx & 31;
        const float* qr = row + h * 64;
        const float* kr = qr + 1024;
        const float* vr = qr + 2048;
        float inv = __expf(-(float)j * 0.28782313662f);  // 10000^(-j/32)
        float ang = (float)t * inv;
        float s, c;
        __sincosf(ang, &s, &c);
        float q1 = qr[j], q2 = qr[j + 32];
        float k1 = kr[j], k2 = kr[j + 32];
        size_t base = ((size_t)h * T + t) * 64;
        qb[base + j]      = (bf16)(q1 * c - q2 * s);
        qb[base + j + 32] = (bf16)(q2 * c + q1 * s);
        kb[base + j]      = (bf16)(k1 * c - k2 * s);
        kb[base + j + 32] = (bf16)(k2 * c + k1 * s);
        size_t vb = (size_t)h * 64 * T;
        vt[vb + (size_t)j * T + t]        = (bf16)vr[j];
        vt[vb + (size_t)(j + 32) * T + t] = (bf16)vr[j + 32];
    }
}

// ---------------------------------------------------------------------------
// Flash attention: one wave per (16-query tile, head). Online softmax over
// 32-key chunks. S via 4 WMMAs, P re-laid out via LDS, PV via 4 WMMAs.
// ---------------------------------------------------------------------------
__global__ void k_attn(const bf16* __restrict__ qb, const bf16* __restrict__ kb,
                       const bf16* __restrict__ vt, float* __restrict__ y, int T) {
    int qt = blockIdx.x, h = blockIdx.y;
    int lane = threadIdx.x & 31;
    const bf16* Q = qb + (size_t)h * T * 64;
    const bf16* K = kb + (size_t)h * T * 64;
    const bf16* V = vt + (size_t)h * 64 * T;   // [d][t]

    v16bf qa0 = load_A_bf16(Q, 64, qt * 16, 0);
    v16bf qa1 = load_A_bf16(Q, 64, qt * 16, 32);

    v8f o[4] = {{}, {}, {}, {}};
    float mr[8], lr[8];
#pragma unroll
    for (int r = 0; r < 8; r++) { mr[r] = -1e30f; lr[r] = 0.f; }

    __shared__ bf16 Pl[16 * 32];
    int n = lane & 15, hi = lane >> 4;
    int nchunks = qt / 2 + 1;

    for (int ch = 0; ch < nchunks; ch++) {
        int k0 = ch * 32;
        v8f s0 = {}, s1 = {};
        {
            v16bf b0 = load_BT_bf16(K, 64, k0, 0);
            v16bf b1 = load_BT_bf16(K, 64, k0, 32);
            s0 = wmma_bf16(qa0, b0, s0);
            s0 = wmma_bf16(qa1, b1, s0);
            v16bf b2 = load_BT_bf16(K, 64, k0 + 16, 0);
            v16bf b3 = load_BT_bf16(K, 64, k0 + 16, 32);
            s1 = wmma_bf16(qa0, b2, s1);
            s1 = wmma_bf16(qa1, b3, s1);
        }
#pragma unroll
        for (int r = 0; r < 8; r++) {
            int qg = qt * 16 + hi * 8 + r;
            float v0 = s0[r] * 0.125f;
            float v1 = s1[r] * 0.125f;
            if (k0 + n > qg)      v0 = -1e30f;
            if (k0 + 16 + n > qg) v1 = -1e30f;
            float mx = fmaxf(v0, v1);
#pragma unroll
            for (int off = 8; off > 0; off >>= 1)
                mx = fmaxf(mx, __shfl_xor(mx, off, 32));
            float mn = fmaxf(mr[r], mx);
            float corr = __expf(mr[r] - mn);
            mr[r] = mn;
            float p0 = __expf(v0 - mn), p1 = __expf(v1 - mn);
            float rs = p0 + p1;
#pragma unroll
            for (int off = 8; off > 0; off >>= 1)
                rs += __shfl_xor(rs, off, 32);
            lr[r] = lr[r] * corr + rs;
#pragma unroll
            for (int dt = 0; dt < 4; dt++) o[dt][r] *= corr;
            Pl[(hi * 8 + r) * 32 + n]      = (bf16)p0;
            Pl[(hi * 8 + r) * 32 + 16 + n] = (bf16)p1;
        }
        __syncthreads();
        v16bf pa = load_A_lds_bf16(Pl, 32, 0);
#pragma unroll
        for (int dt = 0; dt < 4; dt++) {
            v16bf vb = load_BT_bf16(V, T, dt * 16, k0);
            o[dt] = wmma_bf16(pa, vb, o[dt]);
        }
        __syncthreads();
    }
#pragma unroll
    for (int dt = 0; dt < 4; dt++) {
#pragma unroll
        for (int r = 0; r < 8; r++) {
            int m = qt * 16 + hi * 8 + r;
            y[(size_t)m * 1024 + h * 64 + dt * 16 + n] = o[dt][r] / lr[r];
        }
    }
}

// ---------------------------------------------------------------------------
// Router: logits = xn @ router_w^T, top-2 softmax -> dense gates [N,8]
// ---------------------------------------------------------------------------
__global__ void k_router(const float* __restrict__ xn, const float* __restrict__ rw,
                         float* __restrict__ gates, int C) {
    int t = blockIdx.x;
    int e = threadIdx.y;
    int lane = threadIdx.x;
    const float* xr = xn + (size_t)t * C;
    const float* wr = rw + (size_t)e * C;
    float acc = 0.f;
    for (int i = lane; i < C; i += 32) acc += xr[i] * wr[i];
    acc = wave_red_sum(acc);
    __shared__ float lg[8];
    if (lane == 0) lg[e] = acc;
    __syncthreads();
    if (threadIdx.x == 0 && threadIdx.y == 0) {
        int i1 = 0; float v1 = -1e30f;
        for (int k = 0; k < 8; k++) if (lg[k] > v1) { v1 = lg[k]; i1 = k; }
        int i2 = 0; float v2 = -1e30f;
        for (int k = 0; k < 8; k++) if (k != i1 && lg[k] > v2) { v2 = lg[k]; i2 = k; }
        float e2 = __expf(v2 - v1);
        float inv = 1.f / (1.f + e2);
        for (int k = 0; k < 8; k++) gates[(size_t)t * 8 + k] = 0.f;
        gates[(size_t)t * 8 + i1] = inv;
        gates[(size_t)t * 8 + i2] = e2 * inv;
    }
}

// ---------------------------------------------------------------------------
// MoE gate/up fused: h = silu(X@Wg^T) * (X@Wu^T). TDM-staged A, 16x64 per wave.
// ---------------------------------------------------------------------------
__global__ void k_moe_gu(const float* __restrict__ X, const float* __restrict__ Wg,
                         const float* __restrict__ Wu, bf16* __restrict__ H,
                         int M, int N, int K) {
    __shared__ float Ash[16 * KB];
    int m0 = blockIdx.x * 16;
    int n0 = (blockIdx.y * blockDim.y + threadIdx.y) * 64;
    v8f g[4] = {{}, {}, {}, {}};
    v8f u[4] = {{}, {}, {}, {}};
    for (int kc = 0; kc < K; kc += KB) {
        if (threadIdx.y == 0)
            tdm_load_tile(Ash, X + (size_t)m0 * K + kc, 2u, KB, 16u,
                          (unsigned long long)K);
        __syncthreads();
        for (int k0 = 0; k0 < KB; k0 += 32) {
            __builtin_prefetch(Wg + (size_t)n0 * K + kc + k0 + 64, 0, 1);
            v16bf a = load_A_lds_f32(Ash, KB, k0);
#pragma unroll
            for (int j = 0; j < 4; j++) {
                v16bf bg = load_BT_f32(Wg, K, n0 + j * 16, kc + k0);
                g[j] = wmma_bf16(a, bg, g[j]);
                v16bf bu = load_BT_f32(Wu, K, n0 + j * 16, kc + k0);
                u[j] = wmma_bf16(a, bu, u[j]);
            }
        }
        __syncthreads();
    }
    int lane = threadIdx.x & 31;
    int cn = lane & 15, mh = (lane >> 4) << 3;
#pragma unroll
    for (int j = 0; j < 4; j++) {
        int nn = n0 + j * 16 + cn;
#pragma unroll
        for (int r = 0; r < 8; r++) {
            float gv = g[j][r];
            float hv = gv / (1.f + __expf(-gv)) * u[j][r];   // silu(g) * u
            H[(size_t)(m0 + mh + r) * N + nn] = (bf16)hv;
        }
    }
}

// ---------------------------------------------------------------------------
// MoE down-proj with gate: Out[m,n] += gates[m,e] * (H @ Wd^T)[m,n]
// 32-row M-tile: each B-fragment load feeds 2 wmmas.
// ---------------------------------------------------------------------------
__global__ void k_moe_down(const bf16* __restrict__ H, const float* __restrict__ Wd,
                           const float* __restrict__ gates, int e,
                           float* __restrict__ Out, int M, int N, int K) {
    __shared__ bf16 Ash[32 * KB];
    int m0 = blockIdx.x * 32;
    int n0 = (blockIdx.y * blockDim.y + threadIdx.y) * 64;
    v8f acc0[4] = {{}, {}, {}, {}};
    v8f acc1[4] = {{}, {}, {}, {}};
    for (int kc = 0; kc < K; kc += KB) {
        if (threadIdx.y == 0)
            tdm_load_tile(Ash, H + (size_t)m0 * K + kc, 1u, KB, 32u,
                          (unsigned long long)K);
        __syncthreads();
        for (int k0 = 0; k0 < KB; k0 += 32) {
            __builtin_prefetch(Wd + (size_t)n0 * K + kc + k0 + 64, 0, 1);
            v16bf a0 = load_A_lds_bf16(Ash, KB, k0);
            v16bf a1 = load_A_lds_bf16(Ash + 16 * KB, KB, k0);
#pragma unroll
            for (int j = 0; j < 4; j++) {
                v16bf b = load_BT_f32(Wd, K, n0 + j * 16, kc + k0);
                acc0[j] = wmma_bf16(a0, b, acc0[j]);
                acc1[j] = wmma_bf16(a1, b, acc1[j]);
            }
        }
        __syncthreads();
    }
    int lane = threadIdx.x & 31;
    int cn = lane & 15, mh = (lane >> 4) << 3;
    float gt0[8], gt1[8];
#pragma unroll
    for (int r = 0; r < 8; r++) {
        gt0[r] = gates[(size_t)(m0 + mh + r) * 8 + e];
        gt1[r] = gates[(size_t)(m0 + 16 + mh + r) * 8 + e];
    }
#pragma unroll
    for (int j = 0; j < 4; j++) {
        int nn = n0 + j * 16 + cn;
        float old[8];
#pragma unroll
        for (int r = 0; r < 8; r++) old[r] = Out[(size_t)(m0 + mh + r) * N + nn];
#pragma unroll
        for (int r = 0; r < 8; r++)
            Out[(size_t)(m0 + mh + r) * N + nn] = old[r] + gt0[r] * acc0[j][r];
#pragma unroll
        for (int r = 0; r < 8; r++) old[r] = Out[(size_t)(m0 + 16 + mh + r) * N + nn];
#pragma unroll
        for (int r = 0; r < 8; r++)
            Out[(size_t)(m0 + 16 + mh + r) * N + nn] = old[r] + gt1[r] * acc1[j][r];
    }
}

__global__ void k_copy(const float* __restrict__ src, float* __restrict__ dst, int n) {
    int i = blockIdx.x * blockDim.x + threadIdx.x;
    if (i < n) dst[i] = src[i];
}

// ---------------------------------------------------------------------------
// Launch
// ---------------------------------------------------------------------------
extern "C" void kernel_launch(void* const* d_in, const int* in_sizes, int n_in,
                              void* d_out, int out_size, void* d_ws, size_t ws_size,
                              hipStream_t stream) {
    const int T = 2048, C = 1024, E = 8, Hff = 4096, NH = 16;
    const float* x        = (const float*)d_in[0];
    const float* norm1_w  = (const float*)d_in[1];
    const float* norm2_w  = (const float*)d_in[2];
    const float* qkv_w    = (const float*)d_in[3];
    const float* proj_w   = (const float*)d_in[4];
    const float* router_w = (const float*)d_in[5];
    const float* wg       = (const float*)d_in[6];
    const float* wu       = (const float*)d_in[7];
    const float* wd       = (const float*)d_in[8];
    float* out = (float*)d_out;

    char* ws = (char*)d_ws;
    float* xn1   = (float*)(ws);                       //  8 MiB  [T,C]
    float* qkv   = (float*)(ws + (8ull  << 20));       // 24 MiB  [T,3C]
    bf16*  qb    = (bf16*) (ws + (32ull << 20));       //  4 MiB  [H][T][64]
    bf16*  kb    = (bf16*) (ws + (36ull << 20));       //  4 MiB
    bf16*  vt    = (bf16*) (ws + (40ull << 20));       //  4 MiB  [H][64][T]
    float* yatt  = (float*)(ws + (44ull << 20));       //  8 MiB  [T,C]
    float* x1    = (float*)(ws + (52ull << 20));       //  8 MiB
    float* xn2   = (float*)(ws + (60ull << 20));       //  8 MiB
    float* gates = (float*)(ws + (68ull << 20));       // 64 KiB  [T,8]
    bf16*  hbuf  = (bf16*) (ws + (69ull << 20));       // 16 MiB  [T,Hff]

    dim3 wblk(32, 4);

    k_rmsnorm<<<T, 256, 0, stream>>>(x, norm1_w, xn1, C);
    k_gemm_xwT<<<dim3(T / 32, (3 * C) / 256), wblk, 0, stream>>>(xn1, qkv_w, nullptr, qkv, T, 3 * C, C);
    k_rope_pack<<<T, 256, 0, stream>>>(qkv, qb, kb, vt, T);
    k_attn<<<dim3(T / 16, NH), 32, 0, stream>>>(qb, kb, vt, yatt, T);
    k_gemm_xwT<<<dim3(T / 32, C / 256), wblk, 0, stream>>>(yatt, proj_w, x, x1, T, C, C);
    k_rmsnorm<<<T, 256, 0, stream>>>(x1, norm2_w, xn2, C);
    k_router<<<T, dim3(32, 8), 0, stream>>>(xn2, router_w, gates, C);
    k_copy<<<(T * C + 255) / 256, 256, 0, stream>>>(x1, out, T * C);
    for (int e = 0; e < E; e++) {
        const float* wge = wg + (size_t)e * Hff * C;
        const float* wue = wu + (size_t)e * Hff * C;
        const float* wde = wd + (size_t)e * C * Hff;
        k_moe_gu<<<dim3(T / 16, Hff / 256), wblk, 0, stream>>>(xn2, wge, wue, hbuf, T, Hff, C);
        k_moe_down<<<dim3(T / 32, C / 256), wblk, 0, stream>>>(hbuf, wde, gates, e, out, T, C, Hff);
    }
}
